// LinearAttention_74474732912911
// MI455X (gfx1250) — matrix operand order repaired and checked by reading, not compiled, not run on hardware
//
#include <hip/hip_runtime.h>
#include <hip/hip_bf16.h>

typedef __attribute__((ext_vector_type(16))) _Float16 v16h;
typedef __attribute__((ext_vector_type(8)))  _Float16 v8h;
typedef __attribute__((ext_vector_type(8)))  float    v8f;

#define D_MODEL   1024
#define N_HEADS   16
#define FEAT      16
#define HEAD_DIM  64
#define SEQ       1024
#define DF        273     // 1 + 16 + 16*16
#define DFP       288     // DF padded to multiple of 32 (WMMA K)
#define CHUNK     64
#define NCHUNK    (SEQ / CHUNK)
#define ATTN_THREADS 512  // 16 waves

// ---------------------------------------------------------------------------
// WMMA helpers (gfx1250 wave32, 16x16x32 f16 -> f32)
// ---------------------------------------------------------------------------
__device__ __forceinline__ v8f wmma_f16(v16h a, v16h b, v8f c) {
    return __builtin_amdgcn_wmma_f32_16x16x32_f16(false, a, false, b,
                                                  (short)0, c, false, false);
}

// A fragment: 16x32 f16, row-major source with leading dim `ld`.
// lane holds row m0+(lane&15); e=0..7 -> k=k0+8*(lane>>4)+e,
// e=8..15 -> k=k0+16+8*(lane>>4)+(e-8).   (ISA 7.12.2, 16-bit A 16x32)
__device__ __forceinline__ v16h load_a_frag(const _Float16* base, int ld,
                                            int m0, int k0, int lane) {
    const _Float16* p = base + (size_t)(m0 + (lane & 15)) * ld
                             + k0 + 8 * (lane >> 4);
    v8h lo = *(const v8h*)p;
    v8h hi = *(const v8h*)(p + 16);
    v16h a;
#pragma unroll
    for (int e = 0; e < 8; ++e) { a[e] = lo[e]; a[e + 8] = hi[e]; }
    return a;
}

// B fragment: 32x16 f16. Source is B^T row-major: row n holds B[k][n]
// contiguous in k. lane holds col n0+(lane&15); e=0..7 -> k=k0+8*(lane>>4)+e,
// e=8..15 -> k=k0+16+8*(lane>>4)+(e-8)
__device__ __forceinline__ v16h load_bt_frag(const _Float16* base, int ld,
                                             int n0, int k0, int lane) {
    const _Float16* p = base + (size_t)(n0 + (lane & 15)) * ld
                             + k0 + 8 * (lane >> 4);
    v8h lo = *(const v8h*)p;
    v8h hi = *(const v8h*)(p + 16);
    v16h b;
#pragma unroll
    for (int e = 0; e < 8; ++e) { b[e] = lo[e]; b[e + 8] = hi[e]; }
    return b;
}

// ---------------------------------------------------------------------------
// CDNA5 async global->LDS copies (ASYNCcnt-tracked, ISA 15.18.3 op 96/98)
// Generic LDS pointer low 32 bits == LDS byte address (flat aperture rule).
// ---------------------------------------------------------------------------
__device__ __forceinline__ void async_g2l_b128(void* lds, const void* g) {
    unsigned loff = (unsigned)(size_t)lds;
    unsigned long long ga = (unsigned long long)g;
    asm volatile("global_load_async_to_lds_b128 %0, %1, off"
                 :: "v"(loff), "v"(ga) : "memory");
}
__device__ __forceinline__ void async_wait0() {
    asm volatile("s_wait_asynccnt 0x0" ::: "memory");
}

// ---------------------------------------------------------------------------
// f32 -> f16 conversion
// ---------------------------------------------------------------------------
__global__ void cvt_f32_to_f16(const float* __restrict__ in,
                               _Float16* __restrict__ out, int n) {
    for (int i = blockIdx.x * blockDim.x + threadIdx.x; i < n;
         i += gridDim.x * blockDim.x)
        out[i] = (_Float16)in[i];
}

// ---------------------------------------------------------------------------
// Generic WMMA GEMM: C(MxN,f32) = A(MxK,f16) @ B(NxK,f16)^T
// one 16x16 output tile per wave
// ---------------------------------------------------------------------------
__global__ __launch_bounds__(256) void gemm_nt_f16(
    const _Float16* __restrict__ A, const _Float16* __restrict__ B,
    float* __restrict__ C, int M, int N, int K) {
    const int lane = threadIdx.x & 31;
    const int wave = (blockIdx.x * blockDim.x + threadIdx.x) >> 5;
    const int tn = N >> 4;
    const int ntile = (M >> 4) * tn;
    if (wave >= ntile) return;  // wave-uniform
    const int m0 = (wave / tn) << 4;
    const int n0 = (wave % tn) << 4;

    const _Float16* rowA = A + (size_t)(m0 + (lane & 15)) * K;
    const _Float16* rowB = B + (size_t)(n0 + (lane & 15)) * K;

    v8f acc = {};
    for (int k0 = 0; k0 < K; k0 += 32) {
        if (k0 + 32 < K) {
            __builtin_prefetch(rowA + k0 + 32, 0, 3);  // near-cache prefetch
            __builtin_prefetch(rowB + k0 + 32, 0, 3);
        }
        v16h a = load_a_frag(A, K, m0, k0, lane);
        v16h b = load_bt_frag(B, K, n0, k0, lane);
        acc = wmma_f16(a, b, acc);
    }
    const int row = m0 + (lane >> 4) * 8;
    const int col = n0 + (lane & 15);
#pragma unroll
    for (int r = 0; r < 8; ++r)
        C[(size_t)(row + r) * N + col] = acc[r];
}

// ---------------------------------------------------------------------------
// Chunkwise Taylor-feature causal linear attention. One head per workgroup.
// Q,K: (SEQ x N_HEADS*FEAT) f32; V: (SEQ x D_MODEL) f32
// Y:   (SEQ x D_MODEL) f16  (y = num/den)
// ---------------------------------------------------------------------------
__global__ __launch_bounds__(ATTN_THREADS) void taylor_attn(
    const float* __restrict__ Qb, const float* __restrict__ Kb,
    const float* __restrict__ Vb, _Float16* __restrict__ Yb) {
    __shared__ float    qs[CHUNK][FEAT];      // staged q chunk (f32, async)
    __shared__ float    ks[CHUNK][FEAT];      // staged k chunk (f32, async)
    __shared__ float    vStage[CHUNK][CHUNK]; // staged v chunk (f32, async)
    __shared__ _Float16 qf[CHUNK][DFP];       // phi(q), row-major
    __shared__ _Float16 kfr[CHUNK][DFP];      // phi(k), row-major
    __shared__ _Float16 kfc[DFP][CHUNK];      // phi(k), col-major
    __shared__ _Float16 vT[CHUNK][CHUNK];     // v^T  [vcol][pos] (f16)
    __shared__ float    St[CHUNK][DFP];       // state^T [vcol][feat], f32
    __shared__ _Float16 St16[CHUNK][DFP];     // f16 shadow of St for WMMA
    __shared__ float    Am[CHUNK][CHUNK];     // qf @ kf^T (unmasked)
    __shared__ _Float16 Am16[CHUNK][CHUNK];   // causal-masked f16 copy
    __shared__ float    snorm[DFP];           // cumsum of kf over past chunks
    __shared__ float    denp[CHUNK][8];       // den partial sums
    __shared__ float    den[CHUNK];

    const int tid  = threadIdx.x;
    const int lane = tid & 31;
    const int wave = tid >> 5;    // 0..15
    const int h    = blockIdx.x;  // head

    for (int i = tid; i < CHUNK * DFP; i += ATTN_THREADS) {
        (&St[0][0])[i]   = 0.f;
        (&St16[0][0])[i] = (_Float16)0.f;
    }
    for (int i = tid; i < DFP; i += ATTN_THREADS) snorm[i] = 0.f;
    __syncthreads();

    const float c1 = 0.5f;                  // 1/d^(1/4), d=16
    const float c2 = 0.17677669529663687f;  // 1/(sqrt(2)*sqrt(d))

    for (int ch = 0; ch < NCHUNK; ++ch) {
        const int i0 = ch * CHUNK;

        // ---- async stage q, k (64x16 f32) and v (64x64 f32) into LDS ----
        for (int i = tid; i < CHUNK * 4; i += ATTN_THREADS) {   // 4 xb128/row
            int j = i >> 2, part = i & 3;
            async_g2l_b128(&qs[j][part * 4],
                Qb + (size_t)(i0 + j) * (N_HEADS * FEAT) + h * FEAT + part * 4);
            async_g2l_b128(&ks[j][part * 4],
                Kb + (size_t)(i0 + j) * (N_HEADS * FEAT) + h * FEAT + part * 4);
        }
        for (int i = tid; i < CHUNK * 16; i += ATTN_THREADS) {  // 16 xb128/row
            int j = i >> 4, part = i & 15;
            async_g2l_b128(&vStage[j][part * 4],
                Vb + (size_t)(i0 + j) * D_MODEL + h * HEAD_DIM + part * 4);
        }
        async_wait0();
        __syncthreads();

        // ---- v^T in f16 + Taylor feature map ----
        for (int i = tid; i < CHUNK * CHUNK; i += ATTN_THREADS) {
            int j = i >> 6, n = i & 63;
            vT[n][j] = (_Float16)vStage[j][n];
        }
        for (int i = tid; i < CHUNK * DFP; i += ATTN_THREADS) {
            int j = i / DFP, d = i % DFP;
            float qv, kv;
            if (d == 0)        { qv = 1.f; kv = 1.f; }
            else if (d <= 16)  { qv = qs[j][d - 1] * c1; kv = ks[j][d - 1] * c1; }
            else if (d < DF)   { int t = d - 17; int a = t >> 4, b2 = t & 15;
                                 qv = qs[j][a] * qs[j][b2] * c2;
                                 kv = ks[j][a] * ks[j][b2] * c2; }
            else               { qv = 0.f; kv = 0.f; }  // pad 273..287
            qf[j][d] = (_Float16)qv;
            _Float16 kh = (_Float16)kv;
            kfr[j][d] = kh;
            kfc[d][j] = kh;
        }
        __syncthreads();

        // ---- matmul 1: Am = qf @ kf^T   (64x64, K=288), 16 tiles/16 waves --
        {
            int t = wave;
            int m0 = (t >> 2) << 4, n0 = (t & 3) << 4;
            v8f acc = {};
            for (int k0 = 0; k0 < DFP; k0 += 32) {
                v16h a = load_a_frag(&qf[0][0], DFP, m0, k0, lane);
                v16h b = load_bt_frag(&kfr[0][0], DFP, n0, k0, lane);
                acc = wmma_f16(a, b, acc);
            }
            int row = m0 + (lane >> 4) * 8, col = n0 + (lane & 15);
#pragma unroll
            for (int r = 0; r < 8; ++r) Am[row + r][col] = acc[r];
        }
        __syncthreads();

        // ---- causal mask + denominator (8-way split per row) ----
        for (int i = tid; i < CHUNK * CHUNK; i += ATTN_THREADS) {
            int r = i >> 6, c = i & 63;
            Am16[r][c] = (c <= r) ? (_Float16)Am[r][c] : (_Float16)0.f;
        }
        {
            int row = tid >> 3, sub = tid & 7;   // 64 rows x 8 partials
            float s = 0.f;
            for (int d = sub; d < DF; d += 8) s += (float)qf[row][d] * snorm[d];
            for (int j = sub; j <= row; j += 8) s += Am[row][j];
            denp[row][sub] = s;
        }
        __syncthreads();
        if (tid < CHUNK) {
            float s = 1e-12f;  // EPS
#pragma unroll
            for (int p = 0; p < 8; ++p) s += denp[tid][p];
            den[tid] = s;
        }
        __syncthreads();

        // ---- matmul 2: num = qf @ S + mask(Am) @ V ; y = num / den ----
        {
            int t = wave;
            int m0 = (t >> 2) << 4, n0 = (t & 3) << 4;
            v8f acc = {};
            for (int k0 = 0; k0 < DFP; k0 += 32) {        // inter-chunk
                v16h a = load_a_frag(&qf[0][0], DFP, m0, k0, lane);
                v16h b = load_bt_frag(&St16[0][0], DFP, n0, k0, lane);
                acc = wmma_f16(a, b, acc);
            }
            for (int k0 = 0; k0 < CHUNK; k0 += 32) {      // intra-chunk
                v16h a = load_a_frag(&Am16[0][0], CHUNK, m0, k0, lane);
                v16h b = load_bt_frag(&vT[0][0], CHUNK, n0, k0, lane);
                acc = wmma_f16(a, b, acc);
            }
            int row = m0 + (lane >> 4) * 8, col = n0 + (lane & 15);
#pragma unroll
            for (int r = 0; r < 8; ++r) {
                float y = acc[r] / den[row + r];
                Yb[(size_t)(i0 + row + r) * D_MODEL + h * HEAD_DIM + col] =
                    (_Float16)y;
            }
        }

        // ---- matmul 3: St += v^T @ kf (64x288, K=64); snorm += colsum(kf) --
        for (int t = wave; t < 4 * (DFP / 16); t += 16) {  // 72 tiles
            int m0 = (t / (DFP / 16)) << 4;
            int n0 = (t % (DFP / 16)) << 4;
            int row = m0 + (lane >> 4) * 8, col = n0 + (lane & 15);
            v8f acc;
#pragma unroll
            for (int r = 0; r < 8; ++r) acc[r] = St[row + r][col];
            for (int k0 = 0; k0 < CHUNK; k0 += 32) {
                v16h a = load_a_frag(&vT[0][0], CHUNK, m0, k0, lane);
                v16h b = load_bt_frag(&kfc[0][0], CHUNK, n0, k0, lane);
                acc = wmma_f16(a, b, acc);
            }
#pragma unroll
            for (int r = 0; r < 8; ++r) St[row + r][col] = acc[r];
        }
        for (int d = tid; d < DFP; d += ATTN_THREADS) {
            float s = snorm[d];
            for (int j = 0; j < CHUNK; ++j) s += (float)kfr[j][d];
            snorm[d] = s;
        }
        __syncthreads();

        // ---- refresh f16 shadow of state ----
        for (int i = tid; i < CHUNK * DFP; i += ATTN_THREADS)
            (&St16[0][0])[i] = (_Float16)(&St[0][0])[i];
        __syncthreads();
    }
}

// ---------------------------------------------------------------------------
// launch
// ---------------------------------------------------------------------------
extern "C" void kernel_launch(void* const* d_in, const int* in_sizes, int n_in,
                              void* d_out, int out_size, void* d_ws,
                              size_t ws_size, hipStream_t stream) {
    (void)in_sizes; (void)n_in; (void)out_size; (void)ws_size;
    const float* hs = (const float*)d_in[0];
    const float* Wq = (const float*)d_in[1];
    const float* Wk = (const float*)d_in[2];
    const float* Wv = (const float*)d_in[3];
    const float* Wo = (const float*)d_in[4];

    char* ws = (char*)d_ws;
    _Float16* hs16 = (_Float16*)ws;  ws += (size_t)SEQ * D_MODEL * 2;
    _Float16* Wq16 = (_Float16*)ws;  ws += (size_t)N_HEADS * FEAT * D_MODEL * 2;
    _Float16* Wk16 = (_Float16*)ws;  ws += (size_t)N_HEADS * FEAT * D_MODEL * 2;
    _Float16* Wv16 = (_Float16*)ws;  ws += (size_t)D_MODEL * D_MODEL * 2;
    _Float16* Wo16 = (_Float16*)ws;  ws += (size_t)D_MODEL * D_MODEL * 2;
    float*    Qb   = (float*)ws;     ws += (size_t)SEQ * N_HEADS * FEAT * 4;
    float*    Kb   = (float*)ws;     ws += (size_t)SEQ * N_HEADS * FEAT * 4;
    float*    Vb   = (float*)ws;     ws += (size_t)SEQ * D_MODEL * 4;
    _Float16* Yb   = (_Float16*)ws;  ws += (size_t)SEQ * D_MODEL * 2;

    cvt_f32_to_f16<<<1024, 256, 0, stream>>>(hs, hs16, SEQ * D_MODEL);
    cvt_f32_to_f16<<<256, 256, 0, stream>>>(Wq, Wq16, N_HEADS * FEAT * D_MODEL);
    cvt_f32_to_f16<<<256, 256, 0, stream>>>(Wk, Wk16, N_HEADS * FEAT * D_MODEL);
    cvt_f32_to_f16<<<1024, 256, 0, stream>>>(Wv, Wv16, D_MODEL * D_MODEL);
    cvt_f32_to_f16<<<1024, 256, 0, stream>>>(Wo, Wo16, D_MODEL * D_MODEL);

    // Q = hs @ Wq^T, K = hs @ Wk^T, V = hs @ Wv^T
    gemm_nt_f16<<<(SEQ / 16) * (256 / 16) / 8, 256, 0, stream>>>(
        hs16, Wq16, Qb, SEQ, N_HEADS * FEAT, D_MODEL);
    gemm_nt_f16<<<(SEQ / 16) * (256 / 16) / 8, 256, 0, stream>>>(
        hs16, Wk16, Kb, SEQ, N_HEADS * FEAT, D_MODEL);
    gemm_nt_f16<<<(SEQ / 16) * (D_MODEL / 16) / 8, 256, 0, stream>>>(
        hs16, Wv16, Vb, SEQ, D_MODEL, D_MODEL);

    taylor_attn<<<N_HEADS, ATTN_THREADS, 0, stream>>>(Qb, Kb, Vb, Yb);

    // out = y @ Wo^T
    gemm_nt_f16<<<(SEQ / 16) * (D_MODEL / 16) / 8, 256, 0, stream>>>(
        Yb, Wo16, (float*)d_out, SEQ, D_MODEL, D_MODEL);
}
